// RelMultiHeadedAttention_87050397155812
// MI455X (gfx1250) — matrix operand order repaired and checked by reading, not compiled, not run on hardware
//
#include <hip/hip_runtime.h>

// ---------------------------------------------------------------------------
// CDNA5 (gfx1250, wave32) types & helpers: v_wmma_f32_16x16x32_bf16
// ---------------------------------------------------------------------------
typedef __attribute__((ext_vector_type(16))) __bf16 v16bf;
typedef __attribute__((ext_vector_type(8)))  __bf16 v8bf;
typedef __attribute__((ext_vector_type(8)))  float  v8f;
typedef unsigned short ushort_t;

__device__ __forceinline__ unsigned short f2bf(float f) {
    unsigned int u = __float_as_uint(f);
    unsigned int r = (u + 0x7FFFu + ((u >> 16) & 1u)) >> 16;  // RNE
    return (unsigned short)r;
}
__device__ __forceinline__ float bf2f(unsigned short u) {
    return __uint_as_float(((unsigned int)u) << 16);
}

__device__ __forceinline__ v16bf load_frag(const ushort_t* p0, const ushort_t* p1) {
    union { v16bf v; v8bf h[2]; } u;
    u.h[0] = *(const v8bf*)p0;
    u.h[1] = *(const v8bf*)p1;
    return u.v;
}
__device__ __forceinline__ v8f wmma_bf16(v16bf a, v16bf b, v8f c) {
    return __builtin_amdgcn_wmma_f32_16x16x32_bf16(
        false, a, false, b, (short)0, c, false, false);
}
__device__ __forceinline__ v8f vzero8() {
    v8f z = {0.f, 0.f, 0.f, 0.f, 0.f, 0.f, 0.f, 0.f};
    return z;
}

// gfx1250 async global->LDS copy (16B per lane), tracked by ASYNCcnt.
__device__ __forceinline__ void async_copy16(const ushort_t* gsrc, void* ldsdst) {
    unsigned loff = (unsigned)(unsigned long long)ldsdst;  // low 32b = LDS offset
    asm volatile("global_load_async_to_lds_b128 %0, %1, off"
                 :: "v"(loff), "v"(gsrc) : "memory");
}
#define WAIT_ASYNC_0() asm volatile("s_wait_asynccnt 0x0" ::: "memory")
#define WAIT_ASYNC_4() asm volatile("s_wait_asynccnt 0x4" ::: "memory")

// ---------------------------------------------------------------------------
// Precision conversion kernels (run once; tiny vs GEMM cost)
// ---------------------------------------------------------------------------
__global__ __launch_bounds__(256)
void cvt_bf16_kernel(const float4* __restrict__ src, uint2* __restrict__ dst, int n4) {
    int idx = blockIdx.x * 256 + threadIdx.x;
    if (idx < n4) {
        float4 v = src[idx];
        unsigned int p0 = (unsigned)f2bf(v.x) | ((unsigned)f2bf(v.y) << 16);
        unsigned int p1 = (unsigned)f2bf(v.z) | ((unsigned)f2bf(v.w) << 16);
        dst[idx] = make_uint2(p0, p1);
    }
}

// dst(N x K, bf16) = transpose(src(K x N, f32))
__global__ __launch_bounds__(256)
void cvt_t_bf16_kernel(const float* __restrict__ src, ushort_t* __restrict__ dst,
                       int K, int N) {
    __shared__ float t[32][33];
    const int tx = threadIdx.x & 31, ty = threadIdx.x >> 5;
    const int k0 = blockIdx.y * 32, n0 = blockIdx.x * 32;
#pragma unroll
    for (int i = 0; i < 4; ++i)
        t[ty + i * 8][tx] = src[(size_t)(k0 + ty + i * 8) * N + n0 + tx];
    __syncthreads();
#pragma unroll
    for (int i = 0; i < 4; ++i)
        dst[(size_t)(n0 + ty + i * 8) * K + k0 + tx] = f2bf(t[tx][ty + i * 8]);
}

// ---------------------------------------------------------------------------
// Tiled bf16 GEMM: C(MxN) = A(MxK,bf16) * B(KxN) with B given TRANSPOSED
// (BT: NxK bf16). 128x128x32 tile, 256 thr (8 waves), 2x4 WMMA tiles/wave.
// Double-buffered LDS filled by async global->LDS B128 copies.
// ---------------------------------------------------------------------------
#define TM 128
#define TN 128
#define TK 32
#define LDSA 40
#define LDSB 40

template <bool OUT_BF16, bool HAS_RES>
__global__ __launch_bounds__(256)
void gemm_bf16_kernel(const ushort_t* __restrict__ A, const ushort_t* __restrict__ BT,
                      float* __restrict__ Cf, ushort_t* __restrict__ Cb,
                      const float* __restrict__ Res, int M, int N, int K)
{
    __shared__ __align__(32) ushort_t As[2][TM * LDSA];
    __shared__ __align__(32) ushort_t Bs[2][TN * LDSB];

    const int tid  = threadIdx.x;
    const int lane = tid & 31;
    const int wave = tid >> 5;
    const int lrow = lane & 15;
    const int half = lane >> 4;
    const int wm   = wave & 3;
    const int wn   = wave >> 2;
    const int rowBase = blockIdx.y * TM;
    const int colBase = blockIdx.x * TN;

    v8f acc[2][4];
#pragma unroll
    for (int mi = 0; mi < 2; ++mi)
#pragma unroll
        for (int ni = 0; ni < 4; ++ni) acc[mi][ni] = vzero8();

    const int nt = K / TK;
    // stage(t,buf): async-copy 128x32 bf16 A tile and 128x32 bf16 BT tile.
    auto stage = [&](int t, int buf) {
        const int kt = t * TK;
#pragma unroll
        for (int i = 0; i < 2; ++i) {
            int id  = tid + i * 256;       // 512 chunks of 8 bf16
            int row = id >> 2;             // 0..127
            int c   = (id & 3) << 3;       // 0,8,16,24
            async_copy16(A  + (size_t)(rowBase + row) * K + kt + c,
                         &As[buf][row * LDSA + c]);
            async_copy16(BT + (size_t)(colBase + row) * K + kt + c,
                         &Bs[buf][row * LDSB + c]);
        }
    };

    stage(0, 0);
    for (int t = 0; t < nt; ++t) {
        const int cur = t & 1;
        if (t + 1 < nt) { stage(t + 1, cur ^ 1); WAIT_ASYNC_4(); }
        else            { WAIT_ASYNC_0(); }
        __syncthreads();

        v16bf afrag[2], bfrag[4];
#pragma unroll
        for (int mi = 0; mi < 2; ++mi) {
            int am = wm * 32 + mi * 16 + lrow;
            afrag[mi] = load_frag(&As[cur][am * LDSA + 8 * half],
                                  &As[cur][am * LDSA + 16 + 8 * half]);
        }
#pragma unroll
        for (int ni = 0; ni < 4; ++ni) {
            int bn = wn * 64 + ni * 16 + lrow;
            bfrag[ni] = load_frag(&Bs[cur][bn * LDSB + 8 * half],
                                  &Bs[cur][bn * LDSB + 16 + 8 * half]);
        }
#pragma unroll
        for (int mi = 0; mi < 2; ++mi)
#pragma unroll
            for (int ni = 0; ni < 4; ++ni)
                acc[mi][ni] = wmma_bf16(afrag[mi], bfrag[ni], acc[mi][ni]);
        __syncthreads();
    }

    // epilogue; C layout: N = lane%16, M = rr + 8*(lane/16)
#pragma unroll
    for (int mi = 0; mi < 2; ++mi)
#pragma unroll
        for (int ni = 0; ni < 4; ++ni)
#pragma unroll
            for (int rr = 0; rr < 8; ++rr) {
                int g = rowBase + wm * 32 + mi * 16 + half * 8 + rr;
                int c = colBase + wn * 64 + ni * 16 + lrow;
                float v = acc[mi][ni][rr];
                if (HAS_RES) v += Res[(size_t)g * N + c];
                if (OUT_BF16) Cb[(size_t)g * N + c] = f2bf(v);
                else          Cf[(size_t)g * N + c] = v;
            }
}

// ---------------------------------------------------------------------------
// Fused rel-attention per (batch, head, 32-query block); bf16 in / bf16 out.
// rel_shift + causal mask => BD[i,j] = qr_i . rk[j + 1023 - i], j <= i.
// ---------------------------------------------------------------------------
#define AQ  32
#define AJ  64
#define ARK 96
#define ALD 72
#define QLEN 1024
#define NBATCH 4
#define QSTRIDE ((size_t)NBATCH * 3072)

__global__ __launch_bounds__(128)
void relattn_kernel(const ushort_t* __restrict__ qkv, const ushort_t* __restrict__ rk,
                    const float* __restrict__ rwb, const float* __restrict__ rrb,
                    ushort_t* __restrict__ av)
{
    __shared__ __align__(32) ushort_t sQw[AQ * ALD];
    __shared__ __align__(32) ushort_t sQr[AQ * ALD];
    __shared__ __align__(32) ushort_t sKt[AJ * ALD];   // [j][d]
    __shared__ __align__(32) ushort_t sVt[64 * ALD];   // [d][j]
    __shared__ __align__(32) ushort_t sRk[ARK * ALD];  // [mloc][d]; aliased as P
    __shared__ __align__(32) float sSc[AQ * 64];
    __shared__ __align__(32) float sBD[AQ * ARK];
    __shared__ __align__(32) float sM[AQ];
    __shared__ __align__(32) float sL[AQ];
    __shared__ __align__(32) float sC[AQ];

    const int tid  = threadIdx.x;
    const int lane = tid & 31;
    const int wave = tid >> 5;
    const int lrow = lane & 15;
    const int half = lane >> 4;
    const int mi   = wave & 1;
    const int grp  = wave >> 1;

    const int I = blockIdx.x * AQ;
    const int n = blockIdx.y;
    const int b = blockIdx.z;

    const float scale = 0.125f;  // 1/sqrt(64)
    const ushort_t* Qb = qkv + (size_t)b * 3072 + n * 64;
    const ushort_t* Kb = Qb + 1024;
    const ushort_t* Vb = Qb + 2048;

    // ---- Qw = (q+rwb)*scale, Qr = (q+rrb)*scale (bf16 in/out) ----
#pragma unroll
    for (int i = 0; i < 2; ++i) {
        int idx = tid + i * 128;       // 256 chunks of 8
        int row = idx >> 3;
        int c8  = (idx & 7) << 3;
        union { uint4 u; ushort_t s[8]; } q;
        q.u = *(const uint4*)(Qb + (size_t)(I + row) * QSTRIDE + c8);
        float4 bw0 = *(const float4*)(rwb + n * 64 + c8);
        float4 bw1 = *(const float4*)(rwb + n * 64 + c8 + 4);
        float4 br0 = *(const float4*)(rrb + n * 64 + c8);
        float4 br1 = *(const float4*)(rrb + n * 64 + c8 + 4);
        float qf[8], bw[8] = {bw0.x, bw0.y, bw0.z, bw0.w, bw1.x, bw1.y, bw1.z, bw1.w};
        float br[8] = {br0.x, br0.y, br0.z, br0.w, br1.x, br1.y, br1.z, br1.w};
        union { uint4 u; ushort_t s[8]; } ow, orr;
#pragma unroll
        for (int e = 0; e < 8; ++e) {
            qf[e] = bf2f(q.s[e]);
            ow.s[e]  = f2bf((qf[e] + bw[e]) * scale);
            orr.s[e] = f2bf((qf[e] + br[e]) * scale);
        }
        *(uint4*)&sQw[row * ALD + c8] = ow.u;
        *(uint4*)&sQr[row * ALD + c8] = orr.u;
    }
    if (tid < AQ) { sM[tid] = -1e30f; sL[tid] = 0.f; }

    v8f oacc[2];
    oacc[0] = vzero8();
    oacc[1] = vzero8();

    for (int J = 0; J < I + AQ; J += AJ) {
        __syncthreads();  // previous-iteration consumers done before overwrite

        // ---- K tile [j][d]: async copy (64 rows x 64 bf16) ----
#pragma unroll
        for (int i = 0; i < 4; ++i) {
            int id  = tid + i * 128;   // 512 chunks
            int row = id >> 3;
            int c8  = (id & 7) << 3;
            async_copy16(Kb + (size_t)(J + row) * QSTRIDE + c8,
                         &sKt[row * ALD + c8]);
        }
        // ---- rk band [mloc][d]: async copy (96 rows, clamped) ----
        const int mbase = J - I + (QLEN - AQ);
#pragma unroll
        for (int i = 0; i < 6; ++i) {
            int id  = tid + i * 128;   // 768 chunks
            int row = id >> 3;
            int c8  = (id & 7) << 3;
            int msrc = mbase + row;
            if (msrc > QLEN - 1) msrc = QLEN - 1;   // masked positions only
            async_copy16(rk + (size_t)msrc * 1024 + n * 64 + c8,
                         &sRk[row * ALD + c8]);
        }
        // ---- V tile transposed into sVt[d][j] (scatter b16 stores) ----
#pragma unroll
        for (int i = 0; i < 4; ++i) {
            int id  = tid + i * 128;
            int row = id >> 3;
            int c8  = (id & 7) << 3;
            union { uint4 u; ushort_t s[8]; } v;
            v.u = *(const uint4*)(Vb + (size_t)(J + row) * QSTRIDE + c8);
#pragma unroll
            for (int e = 0; e < 8; ++e)
                sVt[(c8 + e) * ALD + row] = v.s[e];
        }
        WAIT_ASYNC_0();
        __syncthreads();

        // ---- scores: AC = Qw.K^T (32x64), BDraw = Qr.Rk^T (32x96) ----
        v16bf aw[2], ar[2];
        const int am = mi * 16 + lrow;
#pragma unroll
        for (int ks = 0; ks < 2; ++ks) {
            aw[ks] = load_frag(&sQw[am * ALD + ks * 32 + 8 * half],
                               &sQw[am * ALD + ks * 32 + 16 + 8 * half]);
            ar[ks] = load_frag(&sQr[am * ALD + ks * 32 + 8 * half],
                               &sQr[am * ALD + ks * 32 + 16 + 8 * half]);
        }
        v8f accS[2], accB[3];
#pragma unroll
        for (int ii = 0; ii < 2; ++ii) {
            accS[ii] = vzero8();
            int ncol = (grp * 2 + ii) * 16 + lrow;
#pragma unroll
            for (int ks = 0; ks < 2; ++ks) {
                v16bf bk = load_frag(&sKt[ncol * ALD + ks * 32 + 8 * half],
                                     &sKt[ncol * ALD + ks * 32 + 16 + 8 * half]);
                accS[ii] = wmma_bf16(aw[ks], bk, accS[ii]);
            }
        }
#pragma unroll
        for (int ii = 0; ii < 3; ++ii) {
            accB[ii] = vzero8();
            int ncol = (grp * 3 + ii) * 16 + lrow;
#pragma unroll
            for (int ks = 0; ks < 2; ++ks) {
                v16bf br = load_frag(&sRk[ncol * ALD + ks * 32 + 8 * half],
                                     &sRk[ncol * ALD + ks * 32 + 16 + 8 * half]);
                accB[ii] = wmma_bf16(ar[ks], br, accB[ii]);
            }
        }
#pragma unroll
        for (int ii = 0; ii < 2; ++ii)
#pragma unroll
            for (int rr = 0; rr < 8; ++rr)
                sSc[(mi * 16 + half * 8 + rr) * 64 + (grp * 2 + ii) * 16 + lrow] =
                    accS[ii][rr];
#pragma unroll
        for (int ii = 0; ii < 3; ++ii)
#pragma unroll
            for (int rr = 0; rr < 8; ++rr)
                sBD[(mi * 16 + half * 8 + rr) * ARK + (grp * 3 + ii) * 16 + lrow] =
                    accB[ii][rr];
        __syncthreads();

        // ---- combine (diagonal rel-shift), mask, online softmax ----
        ushort_t* sP = sRk;  // rk band dead after WMMA phase
        if (tid < AQ) {
            const int i  = tid;
            const int ig = I + i;
            float mold = sM[i];
            float mmax = mold;
            for (int j = 0; j < AJ; ++j) {
                float raw = sSc[i * 64 + j] + sBD[i * ARK + j + (AQ - 1) - i];
                if (J + j > ig) raw = -1e30f;
                sSc[i * 64 + j] = raw;
                mmax = fmaxf(mmax, raw);
            }
            float c = __expf(mold - mmax);
            float sum = 0.f;
            for (int j = 0; j < AJ; ++j) {
                float p = __expf(sSc[i * 64 + j] - mmax);
                sum += p;
                sP[i * ALD + j] = f2bf(p);
            }
            sM[i] = mmax;
            sL[i] = sL[i] * c + sum;
            sC[i] = c;
        }
        __syncthreads();

        // ---- O = O*c + P.V ----
        const ushort_t* sPc = sRk;
        v16bf ap[2];
#pragma unroll
        for (int ks = 0; ks < 2; ++ks)
            ap[ks] = load_frag(&sPc[am * ALD + ks * 32 + 8 * half],
                               &sPc[am * ALD + ks * 32 + 16 + 8 * half]);
        v8f cv = *(const v8f*)&sC[mi * 16 + 8 * half];
#pragma unroll
        for (int ii = 0; ii < 2; ++ii) {
            int dcol = (grp * 2 + ii) * 16 + lrow;
            v8f t = oacc[ii] * cv;
#pragma unroll
            for (int ks = 0; ks < 2; ++ks) {
                v16bf bv = load_frag(&sVt[dcol * ALD + ks * 32 + 8 * half],
                                     &sVt[dcol * ALD + ks * 32 + 16 + 8 * half]);
                t = wmma_bf16(ap[ks], bv, t);
            }
            oacc[ii] = t;
        }
    }

    // ---- normalize, store attn_vec bf16 (qlen, batch, heads*64) ----
    v8f lv = *(const v8f*)&sL[mi * 16 + 8 * half];
#pragma unroll
    for (int ii = 0; ii < 2; ++ii)
#pragma unroll
        for (int rr = 0; rr < 8; ++rr) {
            int g = I + mi * 16 + half * 8 + rr;
            int c = n * 64 + (grp * 2 + ii) * 16 + lrow;
            av[((size_t)g * NBATCH + b) * 1024 + c] = f2bf(oacc[ii][rr] / lv[rr]);
        }
}

// ---------------------------------------------------------------------------
// Launch pipeline
// ---------------------------------------------------------------------------
extern "C" void kernel_launch(void* const* d_in, const int* in_sizes, int n_in,
                              void* d_out, int out_size, void* d_ws, size_t ws_size,
                              hipStream_t stream) {
    (void)in_sizes; (void)n_in; (void)out_size; (void)ws_size;
    const float* w    = (const float*)d_in[0];
    const float* r    = (const float*)d_in[1];
    const float* rwb  = (const float*)d_in[2];
    const float* rrb  = (const float*)d_in[3];
    const float* Wqkv = (const float*)d_in[4];
    const float* Wr   = (const float*)d_in[5];
    const float* Wo   = (const float*)d_in[6];
    float* out = (float*)d_out;

    ushort_t* ws    = (ushort_t*)d_ws;
    ushort_t* wb    = ws;                                 // 4096*1024
    ushort_t* rb    = wb    + (size_t)4096 * 1024;        // 1024*1024
    ushort_t* WqkvT = rb    + (size_t)1024 * 1024;        // 3072*1024
    ushort_t* WrT   = WqkvT + (size_t)3072 * 1024;        // 1024*1024
    ushort_t* WoT   = WrT   + (size_t)1024 * 1024;        // 1024*1024
    ushort_t* qkvb  = WoT   + (size_t)1024 * 1024;        // 4096*3072
    ushort_t* rkb   = qkvb  + (size_t)4096 * 3072;        // 1024*1024
    ushort_t* avb   = rkb   + (size_t)1024 * 1024;        // 4096*1024

    // one-time bf16 conversion / weight transposition
    cvt_bf16_kernel<<<4096, 256, 0, stream>>>((const float4*)w, (uint2*)wb, 4096 * 1024 / 4);
    cvt_bf16_kernel<<<1024, 256, 0, stream>>>((const float4*)r, (uint2*)rb, 1024 * 1024 / 4);
    cvt_t_bf16_kernel<<<dim3(96, 32), 256, 0, stream>>>(Wqkv, WqkvT, 1024, 3072);
    cvt_t_bf16_kernel<<<dim3(32, 32), 256, 0, stream>>>(Wr, WrT, 1024, 1024);
    cvt_t_bf16_kernel<<<dim3(32, 32), 256, 0, stream>>>(Wo, WoT, 1024, 1024);

    // qkv = w @ Wqkv  (bf16 out)
    gemm_bf16_kernel<true, false><<<dim3(3072 / TN, 4096 / TM), 256, 0, stream>>>(
        wb, WqkvT, nullptr, qkvb, nullptr, 4096, 3072, 1024);
    // r_head_k = r @ Wr  (bf16 out)
    gemm_bf16_kernel<true, false><<<dim3(1024 / TN, 1024 / TM), 256, 0, stream>>>(
        rb, WrT, nullptr, rkb, nullptr, 1024, 1024, 1024);
    // fused rel-attention
    relattn_kernel<<<dim3(1024 / AQ, 16, 4), 128, 0, stream>>>(qkvb, rkb, rwb, rrb, avb);
    // out = attn_vec @ Wo + w  (f32 out + residual)
    gemm_bf16_kernel<false, true><<<dim3(1024 / TN, 4096 / TM), 256, 0, stream>>>(
        avb, WoT, out, nullptr, w, 4096, 1024, 1024);
}
